// ViTBlock_65773129171668
// MI455X (gfx1250) — matrix operand order repaired and checked by reading, not compiled, not run on hardware
//
#include <hip/hip_runtime.h>
#include <hip/hip_bf16.h>

typedef __attribute__((ext_vector_type(16))) __bf16 v16bf;
typedef __attribute__((ext_vector_type(8)))  float  v8f;
typedef unsigned int u32;
typedef __attribute__((ext_vector_type(4))) unsigned int u32x4;
typedef __attribute__((ext_vector_type(8))) int i32x8;
typedef __attribute__((ext_vector_type(4))) int i32x4;

#define WMMA_BF16(a, b, c) \
  __builtin_amdgcn_wmma_f32_16x16x32_bf16(false, (a), false, (b), (short)0, (c), false, false)

#if __has_builtin(__builtin_amdgcn_tensor_load_to_lds) && \
    __has_builtin(__builtin_amdgcn_s_wait_tensorcnt)
#define USE_TDM 1
#else
#define USE_TDM 0
#endif

// ---------------------------------------------------------------------------
// Fragment loaders (wave32). Source layout: [row][k] contiguous bf16, row
// chosen by caller per-lane. Per CDNA5 ISA 7.12.2:
//  A (16-bit, 16x32): lanes 0-15 K={0..7,16..23}, lanes 16-31 K={8..15,24..31}
//  B (16-bit, 32x16): lanes 0-15 K={0..7,16..23}, lanes 16-31 K={8..15,24..31}
// ---------------------------------------------------------------------------
__device__ __forceinline__ v16bf frag_a(const __bf16* p_) {
  union { v16bf v; u32 u[8]; } r;
  const u32* p = (const u32*)p_;
  const int o = ((threadIdx.x >> 4) & 1) * 4;
#pragma unroll
  for (int j = 0; j < 4; ++j) { r.u[j] = p[o + j]; r.u[4 + j] = p[8 + o + j]; }
  return r.v;
}
__device__ __forceinline__ v16bf frag_b(const __bf16* p_) {
  union { v16bf v; u32 u[8]; } r;
  const u32* p = (const u32*)p_;
  const int o = ((threadIdx.x >> 4) & 1) * 4;
#pragma unroll
  for (int j = 0; j < 4; ++j) { r.u[j] = p[o + j]; r.u[4 + j] = p[8 + o + j]; }
  return r.v;
}

#if USE_TDM
// ---------------------------------------------------------------------------
// Tensor Data Mover: DMA a 2-D bf16 tile (tile_w x tile_h) from global memory
// (row stride gstride elements) into LDS at byte offset lds_off, with hardware
// row padding: after every (1<<(pad_interval+1)) dwords insert (pad_amount+1)
// dwords of padding.  D# layout per CDNA5 ISA ch.8 (group0/group1; 2-D so
// groups 2/3 are zero).  data_size=1 (2 bytes).  Tracked on TENSORcnt.
// clang-23 builtin form: (u32x4, i32x8, i32x4, i32x4, i32x8, i32 cpol).
// ---------------------------------------------------------------------------
__device__ __forceinline__ void tdm_load_2d(unsigned lds_byte_off,
                                            const void* gptr,
                                            unsigned tile_w, unsigned tile_h,
                                            unsigned gstride,
                                            unsigned pad_interval,
                                            unsigned pad_amount) {
  const unsigned long long ga = (unsigned long long)gptr;
  u32x4 g0;
  g0[0] = 1u;                                   // count=1, user descriptor
  g0[1] = lds_byte_off;                         // lds_addr
  g0[2] = (unsigned)(ga & 0xffffffffu);         // global_addr[31:0]
  g0[3] = (unsigned)((ga >> 32) & 0x01ffffffu)  // global_addr[56:32]
          | 0x80000000u;                        // type=2 ("image")
  i32x8 g1;
  g1[0] = (int)((1u << 16)                      // data_size = 2 bytes
                | (1u << 20)                    // pad_enable
                | (pad_interval << 22)
                | (pad_amount << 25));
  g1[1] = (int)((gstride & 0xffffu) << 16);     // tensor_dim0[15:0]
  g1[2] = (int)(((gstride >> 16) & 0xffffu) | ((tile_h & 0xffffu) << 16)); // dim0 hi | tensor_dim1 lo
  g1[3] = (int)(((tile_h >> 16) & 0xffffu) | ((tile_w & 0xffffu) << 16));  // dim1 hi | tile_dim0
  g1[4] = (int)(tile_h & 0xffffu);              // tile_dim1 (tile_dim2 = 0 -> 2D)
  g1[5] = (int)gstride;                         // tensor_dim0_stride[31:0]
  g1[6] = 0;                                    // stride hi, dim1_stride lo
  g1[7] = 0;
  const i32x4 gz4 = {0, 0, 0, 0};
  const i32x8 gz8 = {0, 0, 0, 0, 0, 0, 0, 0};
  __builtin_amdgcn_tensor_load_to_lds(g0, g1, gz4, gz4, gz8, 0);
}
#endif

__device__ __forceinline__ unsigned lds_off(const void* p) {
  // generic address of an LDS object: addr[31:0] is the LDS byte offset
  return (unsigned)(unsigned long long)p;
}

// ---------------------------------------------------------------------------
// Weight convert + transpose: W (K x N, f32) -> Wt (N x K, bf16)
// ---------------------------------------------------------------------------
__global__ void convt_kernel(const float* __restrict__ W, __bf16* __restrict__ Wt,
                             int K, int N) {
  int id = blockIdx.x * 256 + threadIdx.x;
  if (id >= K * N) return;
  int n = id / K, k = id - n * K;
  Wt[id] = (__bf16)W[(size_t)k * N + n];
}

// ---------------------------------------------------------------------------
// Mean-only LayerNorm over 768, fp32 in -> bf16 out. One row per block.
// ---------------------------------------------------------------------------
__global__ __launch_bounds__(256) void ln_kernel(const float* __restrict__ src,
                                                 const float* __restrict__ w,
                                                 const float* __restrict__ b,
                                                 __bf16* __restrict__ dst) {
  __shared__ float red[8];
  const int tid = threadIdx.x, wave = tid >> 5, lane = tid & 31;
  const size_t row = blockIdx.x;
  const float* x = src + row * 768;
  float v[3];
#pragma unroll
  for (int j = 0; j < 3; ++j) v[j] = x[tid + j * 256];
  float s = v[0] + v[1] + v[2];
#pragma unroll
  for (int off = 16; off; off >>= 1) s += __shfl_xor(s, off, 32);
  if (lane == 0) red[wave] = s;
  __syncthreads();
  float tot = 0.f;
#pragma unroll
  for (int i = 0; i < 8; ++i) tot += red[i];
  const float u = tot * (1.0f / 768.0f);
#pragma unroll
  for (int j = 0; j < 3; ++j) {
    int c = tid + j * 256;
    dst[row * 768 + c] = (__bf16)(w[c] * (v[j] - u) + b[c]);
  }
}

// ---------------------------------------------------------------------------
// bf16 WMMA GEMM: C = A(MxK) * Bt(NxK)^T  (Bt stored N-major, K contiguous)
// 256 threads = 8 waves, 128x128 tile, wave tile 64x32 (4x2 wmma accums),
// double-buffered LDS (TDM-staged, padded stride 40 bf16).
// OMODE: 0 = f32 out (+bias,+residual)       (Wo-proj, W2-proj)
//        1 = bf16 out (+bias,+relu)           (W1-proj)
//        2 = bf16 scatter to [bh][tok][64]    (Q, K)
//        3 = bf16 scatter to [bh][64][tok]    (V transposed)
// ---------------------------------------------------------------------------
template <int OMODE>
__global__ __launch_bounds__(256) void gemm_kernel(
    const __bf16* __restrict__ A, const __bf16* __restrict__ Bt,
    const float* __restrict__ bias, const float* __restrict__ resid,
    float* __restrict__ outF, __bf16* __restrict__ outB,
    int N, int K, int relu) {
  __shared__ __align__(16) __bf16 As[2][128 * 40];
  __shared__ __align__(16) __bf16 Bs[2][128 * 40];

  const int tid = threadIdx.x;
  const int wave = tid >> 5, lane = tid & 31, hf = lane >> 4, ln = lane & 15;
  const int wr = wave >> 2, wc = wave & 3;
  const int mBase = blockIdx.y * 128, nBase = blockIdx.x * 128;

  const __bf16* Ab = A + (size_t)mBase * K;
  const __bf16* Bb = Bt + (size_t)nBase * K;

#if USE_TDM
  const bool issuer = (wave == 0);
  if (issuer) {
    // stage tile 0 for both operands; 128x32 bf16, pad 16dw+4dw -> stride 40
    tdm_load_2d(lds_off(&As[0][0]), Ab, 32, 128, (unsigned)K, 3, 3);
    tdm_load_2d(lds_off(&Bs[0][0]), Bb, 32, 128, (unsigned)K, 3, 3);
    __builtin_amdgcn_s_wait_tensorcnt(0);
  }
#else
  const int c0 = tid, c1 = tid + 256;
  const int r0 = c0 >> 2, o0 = (c0 & 3) * 8;
  const int r1 = c1 >> 2, o1 = (c1 & 3) * 8;
  *(uint4*)(&As[0][r0 * 40 + o0]) = *(const uint4*)(Ab + (size_t)r0 * K + o0);
  *(uint4*)(&As[0][r1 * 40 + o1]) = *(const uint4*)(Ab + (size_t)r1 * K + o1);
  *(uint4*)(&Bs[0][r0 * 40 + o0]) = *(const uint4*)(Bb + (size_t)r0 * K + o0);
  *(uint4*)(&Bs[0][r1 * 40 + o1]) = *(const uint4*)(Bb + (size_t)r1 * K + o1);
#endif
  __syncthreads();

  const v8f vzero = {0.f, 0.f, 0.f, 0.f, 0.f, 0.f, 0.f, 0.f};
  v8f acc[4][2];
#pragma unroll
  for (int s = 0; s < 4; ++s)
#pragma unroll
    for (int t = 0; t < 2; ++t) acc[s][t] = vzero;

  const int KT = K >> 5;
  int buf = 0;
  for (int kt = 0; kt < KT; ++kt) {
    const bool hasNext = (kt + 1 < KT);
    const int k0 = (kt + 1) * 32;
#if USE_TDM
    if (hasNext && issuer) {  // async DMA of next tiles overlaps WMMA below
      const int nb = buf ^ 1;
      tdm_load_2d(lds_off(&As[nb][0]), Ab + k0, 32, 128, (unsigned)K, 3, 3);
      tdm_load_2d(lds_off(&Bs[nb][0]), Bb + k0, 32, 128, (unsigned)K, 3, 3);
    }
#else
    if (hasNext) {
      const int nb = buf ^ 1;
      *(uint4*)(&As[nb][r0 * 40 + o0]) = *(const uint4*)(Ab + (size_t)r0 * K + k0 + o0);
      *(uint4*)(&As[nb][r1 * 40 + o1]) = *(const uint4*)(Ab + (size_t)r1 * K + k0 + o1);
      *(uint4*)(&Bs[nb][r0 * 40 + o0]) = *(const uint4*)(Bb + (size_t)r0 * K + k0 + o0);
      *(uint4*)(&Bs[nb][r1 * 40 + o1]) = *(const uint4*)(Bb + (size_t)r1 * K + k0 + o1);
    }
#endif
    v16bf af[4], bfr[2];
#pragma unroll
    for (int s = 0; s < 4; ++s) af[s] = frag_a(&As[buf][(wr * 64 + s * 16 + ln) * 40]);
#pragma unroll
    for (int t = 0; t < 2; ++t) bfr[t] = frag_b(&Bs[buf][(wc * 32 + t * 16 + ln) * 40]);
#pragma unroll
    for (int s = 0; s < 4; ++s)
#pragma unroll
      for (int t = 0; t < 2; ++t) acc[s][t] = WMMA_BF16(af[s], bfr[t], acc[s][t]);
#if USE_TDM
    if (hasNext && issuer) __builtin_amdgcn_s_wait_tensorcnt(0);
#endif
    __syncthreads();
    buf ^= 1;
  }

#pragma unroll
  for (int s = 0; s < 4; ++s)
#pragma unroll
    for (int t = 0; t < 2; ++t)
#pragma unroll
      for (int i = 0; i < 8; ++i) {
        const int gm = mBase + wr * 64 + s * 16 + i + 8 * hf;
        const int gn = nBase + wc * 32 + t * 16 + ln;
        float val = acc[s][t][i];
        if (OMODE == 0) {
          if (bias) val += bias[gn];
          if (resid) val += resid[(size_t)gm * N + gn];
          outF[(size_t)gm * N + gn] = val;
        } else if (OMODE == 1) {
          if (bias) val += bias[gn];
          if (relu) val = fmaxf(val, 0.0f);
          outB[(size_t)gm * N + gn] = (__bf16)val;
        } else if (OMODE == 2) {
          const int b = gm >> 10, tk = gm & 1023, h = gn >> 6, d = gn & 63;
          outB[((size_t)(b * 12 + h) * 1024 + tk) * 64 + d] = (__bf16)val;
        } else {
          const int b = gm >> 10, tk = gm & 1023, h = gn >> 6, d = gn & 63;
          outB[((size_t)(b * 12 + h) * 64 + d) * 1024 + tk] = (__bf16)val;
        }
      }
}

// ---------------------------------------------------------------------------
// Flash attention. grid = (8 q-tiles, 96 bh). 8 waves; each wave owns 16 q
// rows (Q frags register-resident). K tile 32x64 and Vt tile 64x32 staged in
// double-buffered LDS via TDM; streaming softmax with __shfl_xor reductions.
// ---------------------------------------------------------------------------
__global__ __launch_bounds__(256) void attn_kernel(
    const __bf16* __restrict__ q,   // [bh][1024][64]
    const __bf16* __restrict__ kk,  // [bh][1024][64]
    const __bf16* __restrict__ vt,  // [bh][64][1024]
    __bf16* __restrict__ out) {     // [b][1024][768]
  __shared__ __align__(16) __bf16 Ks[2][32 * 72];
  __shared__ __align__(16) __bf16 Vs[2][64 * 40];
  __shared__ __align__(16) __bf16 Ps[8][16 * 40];

  const int tid = threadIdx.x;
  const int wave = tid >> 5, lane = tid & 31, hf = lane >> 4, ln = lane & 15;
  const int bh = blockIdx.y;
  const int qbase = blockIdx.x * 128 + wave * 16;
  const float SCALE = 0.125f;  // 64^-0.5

  const __bf16* kbase = kk + (size_t)bh * 1024 * 64;
  const __bf16* vbase = vt + (size_t)bh * 64 * 1024;

  const __bf16* qrow = q + ((size_t)bh * 1024 + qbase + ln) * 64;
  const v16bf qa0 = frag_a(qrow);
  const v16bf qa1 = frag_a(qrow + 32);

  const v8f vzero = {0.f, 0.f, 0.f, 0.f, 0.f, 0.f, 0.f, 0.f};
  v8f o[4];
  float m8[8], l8[8];
#pragma unroll
  for (int t = 0; t < 4; ++t) o[t] = vzero;
#pragma unroll
  for (int i = 0; i < 8; ++i) { m8[i] = -1e30f; l8[i] = 0.f; }

#if USE_TDM
  const bool issuer = (wave == 0);
  if (issuer) {
    // K tile: 32 rows x 64 bf16 (32dw + 4dw pad -> stride 72)
    tdm_load_2d(lds_off(&Ks[0][0]), kbase, 64, 32, 64, 4, 3);
    // Vt tile: 64 rows x 32 bf16 (16dw + 4dw pad -> stride 40)
    tdm_load_2d(lds_off(&Vs[0][0]), vbase, 32, 64, 1024, 3, 3);
    __builtin_amdgcn_s_wait_tensorcnt(0);
  }
#else
  const int kr = tid >> 3, ko = (tid & 7) * 8;  // K tile: 32 rows x 64
  const int vr = tid >> 2, vo = (tid & 3) * 8;  // Vt tile: 64 rows x 32
  *(uint4*)(&Ks[0][kr * 72 + ko]) = *(const uint4*)(kbase + (size_t)kr * 64 + ko);
  *(uint4*)(&Vs[0][vr * 40 + vo]) = *(const uint4*)(vbase + (size_t)vr * 1024 + vo);
#endif
  __syncthreads();

  int buf = 0;
  for (int j0 = 0; j0 < 1024; j0 += 32) {
    const bool hasNext = (j0 + 32 < 1024);
#if USE_TDM
    if (hasNext && issuer) {
      const int nb = buf ^ 1;
      tdm_load_2d(lds_off(&Ks[nb][0]), kbase + (size_t)(j0 + 32) * 64, 64, 32, 64, 4, 3);
      tdm_load_2d(lds_off(&Vs[nb][0]), vbase + (j0 + 32), 32, 64, 1024, 3, 3);
    }
#else
    if (hasNext) {
      const int nb = buf ^ 1;
      *(uint4*)(&Ks[nb][kr * 72 + ko]) =
          *(const uint4*)(kbase + (size_t)(j0 + 32 + kr) * 64 + ko);
      *(uint4*)(&Vs[nb][vr * 40 + vo]) =
          *(const uint4*)(vbase + (size_t)vr * 1024 + (j0 + 32) + vo);
    }
#endif

    v8f s0 = vzero, s1 = vzero;
    const v16bf kb0lo = frag_b(&Ks[buf][ln * 72]);
    const v16bf kb0hi = frag_b(&Ks[buf][ln * 72 + 32]);
    const v16bf kb1lo = frag_b(&Ks[buf][(16 + ln) * 72]);
    const v16bf kb1hi = frag_b(&Ks[buf][(16 + ln) * 72 + 32]);
    s0 = WMMA_BF16(qa0, kb0lo, s0);
    s0 = WMMA_BF16(qa1, kb0hi, s0);
    s1 = WMMA_BF16(qa0, kb1lo, s1);
    s1 = WMMA_BF16(qa1, kb1hi, s1);

#pragma unroll
    for (int i = 0; i < 8; ++i) {
      const float a0 = s0[i] * SCALE, a1 = s1[i] * SCALE;
      float mx = fmaxf(a0, a1);
      mx = fmaxf(mx, __shfl_xor(mx, 1, 32));
      mx = fmaxf(mx, __shfl_xor(mx, 2, 32));
      mx = fmaxf(mx, __shfl_xor(mx, 4, 32));
      mx = fmaxf(mx, __shfl_xor(mx, 8, 32));
      const float mn = fmaxf(m8[i], mx);
      const float corr = __expf(m8[i] - mn);
      const float p0 = __expf(a0 - mn), p1 = __expf(a1 - mn);
      float rs = p0 + p1;
      rs += __shfl_xor(rs, 1, 32);
      rs += __shfl_xor(rs, 2, 32);
      rs += __shfl_xor(rs, 4, 32);
      rs += __shfl_xor(rs, 8, 32);
      l8[i] = l8[i] * corr + rs;
      m8[i] = mn;
#pragma unroll
      for (int t = 0; t < 4; ++t) o[t][i] *= corr;
      __bf16* pr = Ps[wave] + (i + 8 * hf) * 40;
      pr[ln] = (__bf16)p0;
      pr[16 + ln] = (__bf16)p1;
    }

    const v16bf pa = frag_a(Ps[wave] + ln * 40);
#pragma unroll
    for (int t = 0; t < 4; ++t) {
      const v16bf vb = frag_b(&Vs[buf][(t * 16 + ln) * 40]);
      o[t] = WMMA_BF16(pa, vb, o[t]);
    }
#if USE_TDM
    if (hasNext && issuer) __builtin_amdgcn_s_wait_tensorcnt(0);
#endif
    __syncthreads();
    buf ^= 1;
  }

  const int b = bh / 12, h = bh % 12;
#pragma unroll
  for (int t = 0; t < 4; ++t)
#pragma unroll
    for (int i = 0; i < 8; ++i) {
      const float inv = 1.0f / l8[i];
      const int token = qbase + i + 8 * hf;
      out[((size_t)b * 1024 + token) * 768 + h * 64 + t * 16 + ln] =
          (__bf16)(o[t][i] * inv);
    }
}

// ---------------------------------------------------------------------------
extern "C" void kernel_launch(void* const* d_in, const int* in_sizes, int n_in,
                              void* d_out, int out_size, void* d_ws, size_t ws_size,
                              hipStream_t stream) {
  (void)in_sizes; (void)n_in; (void)out_size; (void)ws_size;
  const float* x     = (const float*)d_in[0];
  const float* Wq    = (const float*)d_in[1];
  const float* Wk    = (const float*)d_in[2];
  const float* Wv    = (const float*)d_in[3];
  const float* Wo    = (const float*)d_in[4];
  const float* bo    = (const float*)d_in[5];
  const float* ln1w  = (const float*)d_in[6];
  const float* ln1b  = (const float*)d_in[7];
  const float* W1    = (const float*)d_in[8];
  const float* b1    = (const float*)d_in[9];
  const float* W2    = (const float*)d_in[10];
  const float* b2    = (const float*)d_in[11];
  const float* ln2w  = (const float*)d_in[12];
  const float* ln2b  = (const float*)d_in[13];
  float* outp = (float*)d_out;

  const size_t Mtok = 8192;  // 8 x 1024
  char* wsp = (char*)d_ws;
  auto carve = [&](size_t bytes) -> void* {
    void* p = wsp;
    wsp += (bytes + 255) & ~(size_t)255;
    return p;
  };
  __bf16* h1   = (__bf16*)carve(Mtok * 768 * 2);
  __bf16* wqT  = (__bf16*)carve(768 * 768 * 2);
  __bf16* wkT  = (__bf16*)carve(768 * 768 * 2);
  __bf16* wvT  = (__bf16*)carve(768 * 768 * 2);
  __bf16* woT  = (__bf16*)carve(768 * 768 * 2);
  __bf16* w1T  = (__bf16*)carve((size_t)768 * 3072 * 2);
  __bf16* w2T  = (__bf16*)carve((size_t)3072 * 768 * 2);
  __bf16* qb   = (__bf16*)carve(Mtok * 768 * 2);
  __bf16* kb   = (__bf16*)carve(Mtok * 768 * 2);
  __bf16* vTb  = (__bf16*)carve(Mtok * 768 * 2);
  __bf16* attn = (__bf16*)carve(Mtok * 768 * 2);
  float*  x1   = (float*)carve(Mtok * 768 * 4);
  __bf16* h2   = (__bf16*)carve(Mtok * 768 * 2);
  __bf16* act  = (__bf16*)carve(Mtok * 3072 * 2);

  // weight convert + transpose to bf16 [N][K]
  {
    int e = 768 * 768, g = (e + 255) / 256;
    convt_kernel<<<g, 256, 0, stream>>>(Wq, wqT, 768, 768);
    convt_kernel<<<g, 256, 0, stream>>>(Wk, wkT, 768, 768);
    convt_kernel<<<g, 256, 0, stream>>>(Wv, wvT, 768, 768);
    convt_kernel<<<g, 256, 0, stream>>>(Wo, woT, 768, 768);
    int e2 = 768 * 3072, g2 = (e2 + 255) / 256;
    convt_kernel<<<g2, 256, 0, stream>>>(W1, w1T, 768, 3072);
    convt_kernel<<<g2, 256, 0, stream>>>(W2, w2T, 3072, 768);
  }

  // LN1
  ln_kernel<<<(int)Mtok, 256, 0, stream>>>(x, ln1w, ln1b, h1);

  // Q, K, V projections (V written transposed)
  {
    dim3 g(768 / 128, Mtok / 128);
    gemm_kernel<2><<<g, 256, 0, stream>>>(h1, wqT, nullptr, nullptr, nullptr, qb, 768, 768, 0);
    gemm_kernel<2><<<g, 256, 0, stream>>>(h1, wkT, nullptr, nullptr, nullptr, kb, 768, 768, 0);
    gemm_kernel<3><<<g, 256, 0, stream>>>(h1, wvT, nullptr, nullptr, nullptr, vTb, 768, 768, 0);
  }

  // flash attention
  attn_kernel<<<dim3(8, 96), 256, 0, stream>>>(qb, kb, vTb, attn);

  // out-proj + bias + residual -> x1 (f32)
  {
    dim3 g(768 / 128, Mtok / 128);
    gemm_kernel<0><<<g, 256, 0, stream>>>(attn, woT, bo, x, x1, nullptr, 768, 768, 0);
  }

  // LN2
  ln_kernel<<<(int)Mtok, 256, 0, stream>>>(x1, ln2w, ln2b, h2);

  // FFN up + ReLU -> act (bf16)
  {
    dim3 g(3072 / 128, Mtok / 128);
    gemm_kernel<1><<<g, 256, 0, stream>>>(h2, w1T, b1, nullptr, nullptr, act, 3072, 768, 1);
  }

  // FFN down + bias + residual -> out (f32)
  {
    dim3 g(768 / 128, Mtok / 128);
    gemm_kernel<0><<<g, 256, 0, stream>>>(act, w2T, b2, x1, outp, nullptr, 768, 3072, 0);
  }
}
